// RSSM_6047313953288
// MI455X (gfx1250) — matrix operand order repaired and checked by reading, not compiled, not run on hardware
//
#include <hip/hip_runtime.h>

typedef unsigned short u16;
typedef unsigned int   u32;

typedef __attribute__((ext_vector_type(16))) __bf16 v16bf;
typedef __attribute__((ext_vector_type(8)))  float  v8f;

#define T_STEPS 64
#define B_N     512
#define S_N     64
#define A_N     6
#define E_N     1024
#define D_N     600
#define N_N     400
#define KX      96        // padded S+A (70 -> 96, 3 k-steps)
#define DP      608       // padded D   (600 -> 608, 19 k-steps)
#define NP      416       // padded N   (400 -> 416, 13 k-steps)
#define DEP     1632      // padded D+E (608 + 1024, 51 k-steps)
#define OUTW    (6 * S_N + D_N)   // 984
#define MROWS   32        // batch rows per workgroup (2 x 16 M-tiles)

// ---------------- device helpers ----------------

union FragAB { v16bf v; uint4 q[2]; };

__device__ __forceinline__ u16 f2bf(float f) {
  u32 u = __float_as_uint(f);
  u32 r = u + 0x7FFFu + ((u >> 16) & 1u);   // round-to-nearest-even
  return (u16)(r >> 16);
}
__device__ __forceinline__ float eluf(float x)      { return x > 0.f ? x : __expf(x) - 1.f; }
__device__ __forceinline__ float sigmoidf_(float x) { return 1.f / (1.f + __expf(-x)); }
__device__ __forceinline__ float softplusf_(float x){ return x > 30.f ? x : log1pf(__expf(x)); }

// A fragment (16x32 bf16) from row-major panel, stride in elements.
// ISA 7.12.2: lane m=lane&15 holds row m; halves 0..7 -> K=hi*8+0..7,
// halves 8..15 -> K=16+hi*8+0..7  => two aligned 16B loads.
__device__ __forceinline__ void loadA16(FragAB& f, const u16* base, int stride,
                                        int m, int hi, int kb) {
  const u16* p = base + m * stride + kb + hi * 8;
  f.q[0] = *(const uint4*)(p);
  f.q[1] = *(const uint4*)(p + 16);
}
// B fragment (32x16 bf16) from column-major weight panel W[ncol][Kp]:
// lane n=lane&15 holds column n; halves 0..7 -> K=hi*8+0..7,
// halves 8..15 -> K=16+hi*8+0..7 => two 16B loads.
__device__ __forceinline__ void loadB16(FragAB& f, const u16* W, int Kp,
                                        int ncol, int hi, int kb) {
  const u16* p = W + (size_t)ncol * Kp + kb + hi * 8;
  f.q[0] = *(const uint4*)(p);
  f.q[1] = *(const uint4*)(p + 16);
}
__device__ __forceinline__ v8f wmma_bf(const FragAB& a, const FragAB& b, v8f c) {
  return __builtin_amdgcn_wmma_f32_16x16x32_bf16(false, a.v, false, b.v,
                                                 (short)0, c, false, false);
}

// ---------------- one-time prep ----------------

__global__ void zerok(u32* p, long n) {
  long i = (long)blockIdx.x * blockDim.x + threadIdx.x;
  if (i < n) p[i] = 0u;
}

// dst[n][Kp] (bf16 col-major panel) = src[k][n] (fp32 row-major, row stride srcStride)
__global__ void wconv(const float* __restrict__ src, int srcStride, int colOff,
                      int K, int Nn, u16* __restrict__ dst, int Kp, int kDst0) {
  int i = blockIdx.x * blockDim.x + threadIdx.x;
  if (i >= K * Nn) return;
  int k = i / Nn, n = i - k * Nn;
  dst[(size_t)n * Kp + kDst0 + k] = f2bf(src[(size_t)k * srcStride + colOff + n]);
}

__global__ void biaspad(const float* __restrict__ src, float* __restrict__ dst,
                        int n, int np) {
  int i = blockIdx.x * blockDim.x + threadIdx.x;
  if (i < np) dst[i] = (i < n) ? src[i] : 0.f;
}

// ---------------- per-step activation panels (bf16, padded) ----------------

__global__ void prep_step(const float* __restrict__ stoch_src,
                          const float* __restrict__ deter_src, int dstride,
                          const float* __restrict__ action_t,
                          const float* __restrict__ nonterm_t,
                          const float* __restrict__ obs_t,
                          u16* __restrict__ x_bf, u16* __restrict__ dh_bf,
                          u16* __restrict__ obs_bf) {
  const int PER = KX + DP + E_N;   // 1728
  int i = blockIdx.x * blockDim.x + threadIdx.x;
  if (i >= B_N * PER) return;
  int row = i / PER, j = i - row * PER;
  if (j < KX) {
    float nt = nonterm_t[row];
    float v = 0.f;
    if (j < S_N)            v = stoch_src[(size_t)row * S_N + j] * nt;
    else if (j < S_N + A_N) v = action_t[(size_t)row * A_N + (j - S_N)] * nt;
    x_bf[(size_t)row * KX + j] = f2bf(v);
  } else if (j < KX + DP) {
    int k = j - KX;
    float nt = nonterm_t[row];
    float v = (k < D_N) ? deter_src[(size_t)row * dstride + k] * nt : 0.f;
    dh_bf[(size_t)row * DP + k] = f2bf(v);
  } else {
    int k = j - KX - DP;
    obs_bf[(size_t)row * E_N + k] = f2bf(obs_t[(size_t)row * E_N + k]);
  }
}

// ---------------- step kernels (grid = B/32, block = 256 = 8 wave32) ----------------

__global__ __launch_bounds__(256)
void gru_kernel(const u16* __restrict__ x_bf, const u16* __restrict__ dh_bf,
                const float* __restrict__ deter_src, int dstride,
                const float* __restrict__ nonterm_t,
                const u16* __restrict__ WsaT, const float* __restrict__ bsa_p,
                const u16* __restrict__ WihT, const float* __restrict__ bih_p,
                const u16* __restrict__ WhhT, const float* __restrict__ bhh_p,
                float* __restrict__ deter_ws, u16* __restrict__ deter_bf,
                float* __restrict__ out_t)
{
  __shared__ alignas(16) u16 sa_lds[MROWS * DP];   // 38 KB
  __shared__ float nt_lds[MROWS];

  const int tid = threadIdx.x;
  const int m0  = blockIdx.x * MROWS;
  if (tid < MROWS) nt_lds[tid] = nonterm_t[m0 + tid];
  __syncthreads();

  const int wave = __builtin_amdgcn_readfirstlane(tid) >> 5;   // wave-uniform
  const int lane = tid & 31, n16 = lane & 15, hi = lane >> 4;

  // phase 1: sa = elu(x @ W_sa + b_sa) for 32 rows -> sa_lds (bf16)
  for (int jt = wave; jt < DP / 16; jt += 8) {
    const int ncol = jt * 16 + n16;
    v8f acc0 = {}, acc1 = {};
    for (int kb = 0; kb < KX; kb += 32) {
      FragAB a0, a1, b;
      loadA16(a0, x_bf + (size_t)m0 * KX, KX, n16, hi, kb);
      loadA16(a1, x_bf + (size_t)(m0 + 16) * KX, KX, n16, hi, kb);
      loadB16(b, WsaT, KX, ncol, hi, kb);
      acc0 = wmma_bf(a0, b, acc0);
      acc1 = wmma_bf(a1, b, acc1);
    }
    const float bias = bsa_p[ncol];
    for (int v = 0; v < 8; ++v) {
      int m = hi * 8 + v;
      sa_lds[m * DP + ncol]        = f2bf(eluf(acc0[v] + bias));
      sa_lds[(m + 16) * DP + ncol] = f2bf(eluf(acc1[v] + bias));
    }
  }
  __syncthreads();

  // phase 2: six gate GEMMs fused; B-fragments amortized over 2 M-tiles
  for (int jt = wave; jt < DP / 16; jt += 8) {
    const int ncol = jt * 16 + n16;
    // prefetch next tile's weight columns (hint -> global_prefetch_b8)
    {
      int pn = ncol + 8 * 16;
      if (pn < DP) {
        __builtin_prefetch(WihT + (size_t)pn * DP, 0, 1);
        __builtin_prefetch(WhhT + (size_t)pn * DP, 0, 1);
      }
    }
    v8f ir0 = {}, iz0 = {}, in0 = {}, hr0 = {}, hz0 = {}, hn0 = {};
    v8f ir1 = {}, iz1 = {}, in1 = {}, hr1 = {}, hz1 = {}, hn1 = {};
    for (int kb = 0; kb < DP; kb += 32) {
      FragAB fa0, fa1, fh0, fh1, fb;
      loadA16(fa0, sa_lds, DP, n16, hi, kb);
      loadA16(fa1, sa_lds + 16 * DP, DP, n16, hi, kb);
      loadA16(fh0, dh_bf + (size_t)m0 * DP, DP, n16, hi, kb);
      loadA16(fh1, dh_bf + (size_t)(m0 + 16) * DP, DP, n16, hi, kb);
      loadB16(fb, WihT,                       DP, ncol, hi, kb); ir0 = wmma_bf(fa0, fb, ir0); ir1 = wmma_bf(fa1, fb, ir1);
      loadB16(fb, WihT + (size_t)1 * DP * DP, DP, ncol, hi, kb); iz0 = wmma_bf(fa0, fb, iz0); iz1 = wmma_bf(fa1, fb, iz1);
      loadB16(fb, WihT + (size_t)2 * DP * DP, DP, ncol, hi, kb); in0 = wmma_bf(fa0, fb, in0); in1 = wmma_bf(fa1, fb, in1);
      loadB16(fb, WhhT,                       DP, ncol, hi, kb); hr0 = wmma_bf(fh0, fb, hr0); hr1 = wmma_bf(fh1, fb, hr1);
      loadB16(fb, WhhT + (size_t)1 * DP * DP, DP, ncol, hi, kb); hz0 = wmma_bf(fh0, fb, hz0); hz1 = wmma_bf(fh1, fb, hz1);
      loadB16(fb, WhhT + (size_t)2 * DP * DP, DP, ncol, hi, kb); hn0 = wmma_bf(fh0, fb, hn0); hn1 = wmma_bf(fh1, fb, hn1);
    }
    // branch-free epilogue (padded biases); only the out_t store is predicated
    const bool valid = ncol < D_N;
    const int  colc  = valid ? ncol : (D_N - 1);
    const float bir = bih_p[ncol], biz = bih_p[DP + ncol], bin = bih_p[2 * DP + ncol];
    const float bhr = bhh_p[ncol], bhz = bhh_p[DP + ncol], bhn = bhh_p[2 * DP + ncol];
    auto epi = [&](const v8f& ir, const v8f& iz, const v8f& inn,
                   const v8f& hr, const v8f& hz, const v8f& hn, int mbase) {
      for (int v = 0; v < 8; ++v) {
        int m = mbase + hi * 8 + v;
        int row = m0 + m;
        float r  = sigmoidf_(ir[v] + bir + hr[v] + bhr);
        float z  = sigmoidf_(iz[v] + biz + hz[v] + bhz);
        float nn = tanhf(inn[v] + bin + r * (hn[v] + bhn));
        float h  = deter_src[(size_t)row * dstride + colc] * nt_lds[m];
        float dn = (1.f - z) * nn + z * h;
        dn = valid ? dn : 0.f;
        deter_ws[(size_t)row * DP + ncol] = dn;
        deter_bf[(size_t)row * DP + ncol] = f2bf(dn);
        if (valid) out_t[(size_t)row * OUTW + 6 * S_N + ncol] = dn;
      }
    };
    epi(ir0, iz0, in0, hr0, hz0, hn0, 0);
    epi(ir1, iz1, in1, hr1, hz1, hn1, 16);
  }
}

// Shared two-layer head: h = elu(A @ W1 + b1); o = h @ W2 + b2; reparameterize.
__global__ __launch_bounds__(256)
void prior_kernel(const u16* __restrict__ deter_bf,
                  const u16* __restrict__ Wp1T, const float* __restrict__ bp1_p,
                  const u16* __restrict__ Wp2T, const float* __restrict__ bp2,
                  const float* __restrict__ eps0, float* __restrict__ out_t)
{
  __shared__ alignas(16) u16 h_lds[MROWS * NP];   // 26 KB
  __shared__ float o_lds[MROWS * 128];            // 16 KB
  const int tid = threadIdx.x, m0 = blockIdx.x * MROWS;
  const int wave = __builtin_amdgcn_readfirstlane(tid) >> 5;
  const int lane = tid & 31, n16 = lane & 15, hi = lane >> 4;

  for (int jt = wave; jt < NP / 16; jt += 8) {
    const int ncol = jt * 16 + n16;
    {
      int pn = ncol + 8 * 16;
      if (pn < NP) __builtin_prefetch(Wp1T + (size_t)pn * DP, 0, 1);
    }
    v8f acc0 = {}, acc1 = {};
    for (int kb = 0; kb < DP; kb += 32) {
      FragAB a0, a1, b;
      loadA16(a0, deter_bf + (size_t)m0 * DP, DP, n16, hi, kb);
      loadA16(a1, deter_bf + (size_t)(m0 + 16) * DP, DP, n16, hi, kb);
      loadB16(b, Wp1T, DP, ncol, hi, kb);
      acc0 = wmma_bf(a0, b, acc0);
      acc1 = wmma_bf(a1, b, acc1);
    }
    const float bias = bp1_p[ncol];   // zero in pad region -> elu(0)=0
    for (int v = 0; v < 8; ++v) {
      int m = hi * 8 + v;
      h_lds[m * NP + ncol]        = f2bf(eluf(acc0[v] + bias));
      h_lds[(m + 16) * NP + ncol] = f2bf(eluf(acc1[v] + bias));
    }
  }
  __syncthreads();

  {
    const int jt = wave;              // 8 tiles, one per wave
    const int ncol = jt * 16 + n16;
    v8f acc0 = {}, acc1 = {};
    for (int kb = 0; kb < NP; kb += 32) {
      FragAB a0, a1, b;
      loadA16(a0, h_lds, NP, n16, hi, kb);
      loadA16(a1, h_lds + 16 * NP, NP, n16, hi, kb);
      loadB16(b, Wp2T, NP, ncol, hi, kb);
      acc0 = wmma_bf(a0, b, acc0);
      acc1 = wmma_bf(a1, b, acc1);
    }
    const float bias = bp2[ncol];
    for (int v = 0; v < 8; ++v) {
      int m = hi * 8 + v;
      o_lds[m * 128 + ncol]        = acc0[v] + bias;
      o_lds[(m + 16) * 128 + ncol] = acc1[v] + bias;
    }
  }
  __syncthreads();

  for (int i = tid; i < MROWS * S_N; i += 256) {
    int m = i / S_N, s = i - m * S_N, row = m0 + m;
    float pm   = o_lds[m * 128 + s];
    float pstd = softplusf_(o_lds[m * 128 + S_N + s]) + 0.1f;
    float pst  = pm + pstd * eps0[(size_t)row * S_N + s];
    float* ob = out_t + (size_t)row * OUTW;
    ob[s]           = pm;
    ob[S_N + s]     = pstd;
    ob[2 * S_N + s] = pst;
  }
}

__global__ __launch_bounds__(256)
void post_kernel(const u16* __restrict__ deter_bf, const u16* __restrict__ obs_bf,
                 const u16* __restrict__ Wq1T, const float* __restrict__ bq1_p,
                 const u16* __restrict__ Wq2T, const float* __restrict__ bq2,
                 const float* __restrict__ eps1, float* __restrict__ out_t,
                 float* __restrict__ stoch_ws)
{
  __shared__ alignas(16) u16 h_lds[MROWS * NP];
  __shared__ float o_lds[MROWS * 128];
  const int tid = threadIdx.x, m0 = blockIdx.x * MROWS;
  const int wave = __builtin_amdgcn_readfirstlane(tid) >> 5;
  const int lane = tid & 31, n16 = lane & 15, hi = lane >> 4;

  // h = elu([deter | obs] @ Wq1 + bq1); K split: [0,DP) deter panel, [DP,DEP) obs panel
  for (int jt = wave; jt < NP / 16; jt += 8) {
    const int ncol = jt * 16 + n16;
    {
      int pn = ncol + 8 * 16;
      if (pn < NP) __builtin_prefetch(Wq1T + (size_t)pn * DEP, 0, 1);
    }
    v8f acc0 = {}, acc1 = {};
    for (int kb = 0; kb < DEP; kb += 32) {
      FragAB a0, a1, b;
      if (kb < DP) {
        loadA16(a0, deter_bf + (size_t)m0 * DP, DP, n16, hi, kb);
        loadA16(a1, deter_bf + (size_t)(m0 + 16) * DP, DP, n16, hi, kb);
      } else {
        loadA16(a0, obs_bf + (size_t)m0 * E_N, E_N, n16, hi, kb - DP);
        loadA16(a1, obs_bf + (size_t)(m0 + 16) * E_N, E_N, n16, hi, kb - DP);
      }
      loadB16(b, Wq1T, DEP, ncol, hi, kb);
      acc0 = wmma_bf(a0, b, acc0);
      acc1 = wmma_bf(a1, b, acc1);
    }
    const float bias = bq1_p[ncol];
    for (int v = 0; v < 8; ++v) {
      int m = hi * 8 + v;
      h_lds[m * NP + ncol]        = f2bf(eluf(acc0[v] + bias));
      h_lds[(m + 16) * NP + ncol] = f2bf(eluf(acc1[v] + bias));
    }
  }
  __syncthreads();

  {
    const int jt = wave;
    const int ncol = jt * 16 + n16;
    v8f acc0 = {}, acc1 = {};
    for (int kb = 0; kb < NP; kb += 32) {
      FragAB a0, a1, b;
      loadA16(a0, h_lds, NP, n16, hi, kb);
      loadA16(a1, h_lds + 16 * NP, NP, n16, hi, kb);
      loadB16(b, Wq2T, NP, ncol, hi, kb);
      acc0 = wmma_bf(a0, b, acc0);
      acc1 = wmma_bf(a1, b, acc1);
    }
    const float bias = bq2[ncol];
    for (int v = 0; v < 8; ++v) {
      int m = hi * 8 + v;
      o_lds[m * 128 + ncol]        = acc0[v] + bias;
      o_lds[(m + 16) * 128 + ncol] = acc1[v] + bias;
    }
  }
  __syncthreads();

  for (int i = tid; i < MROWS * S_N; i += 256) {
    int m = i / S_N, s = i - m * S_N, row = m0 + m;
    float qm   = o_lds[m * 128 + s];
    float qstd = softplusf_(o_lds[m * 128 + S_N + s]) + 0.1f;
    float qst  = qm + qstd * eps1[(size_t)row * S_N + s];
    float* ob = out_t + (size_t)row * OUTW;
    ob[3 * S_N + s] = qm;
    ob[4 * S_N + s] = qstd;
    ob[5 * S_N + s] = qst;
    stoch_ws[(size_t)row * S_N + s] = qst;
  }
}

// ---------------- host ----------------

extern "C" void kernel_launch(void* const* d_in, const int* in_sizes, int n_in,
                              void* d_out, int out_size, void* d_ws, size_t ws_size,
                              hipStream_t stream) {
  (void)in_sizes; (void)n_in; (void)out_size; (void)ws_size;

  const float* obs   = (const float*)d_in[0];
  const float* act   = (const float*)d_in[1];
  const float* nts   = (const float*)d_in[2];
  const float* noise = (const float*)d_in[3];
  const float* init_stoch = (const float*)d_in[4];
  const float* init_deter = (const float*)d_in[5];
  const float* W_sa = (const float*)d_in[6];  const float* b_sa = (const float*)d_in[7];
  const float* W_ih = (const float*)d_in[8];  const float* b_ih = (const float*)d_in[9];
  const float* W_hh = (const float*)d_in[10]; const float* b_hh = (const float*)d_in[11];
  const float* Wp1  = (const float*)d_in[12]; const float* bp1  = (const float*)d_in[13];
  const float* Wp2  = (const float*)d_in[14]; const float* bp2  = (const float*)d_in[15];
  const float* Wq1  = (const float*)d_in[16]; const float* bq1  = (const float*)d_in[17];
  const float* Wq2  = (const float*)d_in[18]; const float* bq2  = (const float*)d_in[19];

  char* base = (char*)d_ws;
  size_t off = 0;
  auto take = [&](size_t bytes) -> char* {
    char* p = base + off;
    off = (off + bytes + 255) & ~(size_t)255;
    return p;
  };

  // bf16 weight panels (zero-padded region)
  u16* WsaT = (u16*)take((size_t)DP * KX * 2);
  u16* WihT = (u16*)take((size_t)3 * DP * DP * 2);
  u16* WhhT = (u16*)take((size_t)3 * DP * DP * 2);
  u16* Wp1T = (u16*)take((size_t)NP * DP * 2);
  u16* Wp2T = (u16*)take((size_t)128 * NP * 2);
  u16* Wq1T = (u16*)take((size_t)NP * DEP * 2);
  u16* Wq2T = (u16*)take((size_t)128 * NP * 2);
  size_t wend = off;                       // weight region end (256-aligned)
  // padded fp32 biases
  float* bsa_p = (float*)take((size_t)DP * 4);
  float* bih_p = (float*)take((size_t)3 * DP * 4);
  float* bhh_p = (float*)take((size_t)3 * DP * 4);
  float* bp1_p = (float*)take((size_t)NP * 4);
  float* bq1_p = (float*)take((size_t)NP * 4);
  // per-step activation panels + recurrent state
  u16*   x_bf     = (u16*)take((size_t)B_N * KX * 2);
  u16*   dh_bf    = (u16*)take((size_t)B_N * DP * 2);
  u16*   obs_bf   = (u16*)take((size_t)B_N * E_N * 2);
  u16*   deter_bf = (u16*)take((size_t)B_N * DP * 2);
  float* stoch_ws = (float*)take((size_t)B_N * S_N * 4);
  float* deter_ws = (float*)take((size_t)B_N * DP * 4);   // padded stride DP

  auto g1 = [](long n) { return dim3((unsigned)((n + 255) / 256)); };

  zerok<<<g1((long)(wend / 4)), 256, 0, stream>>>((u32*)d_ws, (long)(wend / 4));
  wconv<<<g1(70L * 600), 256, 0, stream>>>(W_sa, 600, 0, 70, 600, WsaT, KX, 0);
  for (int g = 0; g < 3; ++g) {
    wconv<<<g1(600L * 600), 256, 0, stream>>>(W_ih, 1800, g * 600, 600, 600,
                                              WihT + (size_t)g * DP * DP, DP, 0);
    wconv<<<g1(600L * 600), 256, 0, stream>>>(W_hh, 1800, g * 600, 600, 600,
                                              WhhT + (size_t)g * DP * DP, DP, 0);
  }
  wconv<<<g1(600L * 400), 256, 0, stream>>>(Wp1, 400, 0, 600, 400, Wp1T, DP, 0);
  wconv<<<g1(400L * 128), 256, 0, stream>>>(Wp2, 128, 0, 400, 128, Wp2T, NP, 0);
  wconv<<<g1(600L * 400), 256, 0, stream>>>(Wq1, 400, 0, 600, 400, Wq1T, DEP, 0);
  wconv<<<g1(1024L * 400), 256, 0, stream>>>(Wq1 + (size_t)600 * 400, 400, 0, 1024, 400,
                                             Wq1T, DEP, DP);
  wconv<<<g1(400L * 128), 256, 0, stream>>>(Wq2, 128, 0, 400, 128, Wq2T, NP, 0);

  biaspad<<<g1(DP), 256, 0, stream>>>(b_sa, bsa_p, 600, DP);
  for (int g = 0; g < 3; ++g) {
    biaspad<<<g1(DP), 256, 0, stream>>>(b_ih + (size_t)g * 600, bih_p + (size_t)g * DP, 600, DP);
    biaspad<<<g1(DP), 256, 0, stream>>>(b_hh + (size_t)g * 600, bhh_p + (size_t)g * DP, 600, DP);
  }
  biaspad<<<g1(NP), 256, 0, stream>>>(bp1, bp1_p, 400, NP);
  biaspad<<<g1(NP), 256, 0, stream>>>(bq1, bq1_p, 400, NP);

  const dim3 grid(B_N / MROWS), block(256);
  const long prep_n = (long)B_N * (KX + DP + E_N);
  for (int t = 0; t < T_STEPS; ++t) {
    const float* stoch_src = (t == 0) ? init_stoch : stoch_ws;
    const float* deter_src = (t == 0) ? init_deter : deter_ws;
    const int dstride      = (t == 0) ? D_N : DP;
    const float* action_t  = act + (size_t)t * B_N * A_N;
    const float* nonterm_t = nts + (size_t)t * B_N;
    const float* obs_t     = obs + (size_t)t * B_N * E_N;
    const float* eps0      = noise + ((size_t)t * 2 + 0) * B_N * S_N;
    const float* eps1      = noise + ((size_t)t * 2 + 1) * B_N * S_N;
    float* out_t           = (float*)d_out + (size_t)t * B_N * OUTW;

    prep_step<<<g1(prep_n), 256, 0, stream>>>(stoch_src, deter_src, dstride,
                                              action_t, nonterm_t, obs_t,
                                              x_bf, dh_bf, obs_bf);
    gru_kernel<<<grid, block, 0, stream>>>(x_bf, dh_bf, deter_src, dstride, nonterm_t,
                                           WsaT, bsa_p, WihT, bih_p, WhhT, bhh_p,
                                           deter_ws, deter_bf, out_t);
    prior_kernel<<<grid, block, 0, stream>>>(deter_bf, Wp1T, bp1_p, Wp2T, bp2, eps0, out_t);
    post_kernel<<<grid, block, 0, stream>>>(deter_bf, obs_bf, Wq1T, bq1_p, Wq2T, bq2,
                                            eps1, out_t, stoch_ws);
  }
}
